// PointNetSetAbstraction_26620207301335
// MI455X (gfx1250) — matrix operand order repaired and checked by reading, not compiled, not run on hardware
//
#include <hip/hip_runtime.h>
#include <hip/hip_bf16.h>

// ---------------------------------------------------------------------------
// PointNet++ Set Abstraction for MI455X (gfx1250, wave32, WMMA)
// Point-major activation layouts => contiguous vector loads for WMMA frags.
// ---------------------------------------------------------------------------

typedef __attribute__((ext_vector_type(16))) __bf16          v16bf;
typedef __attribute__((ext_vector_type(8)))  float           v8f;
typedef __attribute__((ext_vector_type(16))) float           v16f;
typedef __attribute__((ext_vector_type(16))) unsigned short  v16u;

static constexpr int Bc  = 4;
static constexpr int Nc  = 8192;
static constexpr int Sc  = 2048;
static constexpr int Kc  = 32;
static constexpr int Dc  = 64;
static constexpr int NPc = Bc * Sc * Kc;          // 262144 GEMM columns
static constexpr float R2c = 0.2f * 0.2f;

__device__ __forceinline__ unsigned short f2bf(float f) {
    unsigned u = __float_as_uint(f);
    u += 0x7fffu + ((u >> 16) & 1u);              // round-to-nearest-even
    return (unsigned short)(u >> 16);
}

// ---------------------------------------------------------------------------
// 1) Farthest point sampling: one block per batch, points live in registers.
// ---------------------------------------------------------------------------
__global__ __launch_bounds__(256) void fps_kernel(const float* __restrict__ xyz,
                                                  int* __restrict__ fps_idx,
                                                  float* __restrict__ new_xyz) {
    const int b = blockIdx.x;
    const int t = threadIdx.x;
    const float* X = xyz + (size_t)b * Nc * 3;

    __shared__ float rv[256];
    __shared__ int   ri[256];
    __shared__ float cent[3];

    float px[32], py[32], pz[32], dist[32];
#pragma unroll
    for (int j = 0; j < 32; ++j) {
        const int n = j * 256 + t;
        px[j] = X[n * 3 + 0];
        py[j] = X[n * 3 + 1];
        pz[j] = X[n * 3 + 2];
        dist[j] = 1e18f;
    }

    int far = 0;
    for (int s = 0; s < Sc; ++s) {
        if (t == 0) {
            fps_idx[b * Sc + s] = far;
            const float cx = X[far * 3 + 0], cy = X[far * 3 + 1], cz = X[far * 3 + 2];
            cent[0] = cx; cent[1] = cy; cent[2] = cz;
            new_xyz[(b * Sc + s) * 3 + 0] = cx;
            new_xyz[(b * Sc + s) * 3 + 1] = cy;
            new_xyz[(b * Sc + s) * 3 + 2] = cz;
        }
        __syncthreads();
        const float cx = cent[0], cy = cent[1], cz = cent[2];

        float bd = -1.0f; int bi = 0x7fffffff;
#pragma unroll
        for (int j = 0; j < 32; ++j) {
            const float dx = px[j] - cx, dy = py[j] - cy, dz = pz[j] - cz;
            const float d = dx * dx + dy * dy + dz * dz;
            dist[j] = fminf(dist[j], d);
            const int n = j * 256 + t;
            if (dist[j] > bd) { bd = dist[j]; bi = n; }
        }
        rv[t] = bd; ri[t] = bi;
        __syncthreads();
        for (int off = 128; off > 0; off >>= 1) {
            if (t < off) {
                const float ov = rv[t + off]; const int oi = ri[t + off];
                if (ov > rv[t] || (ov == rv[t] && oi < ri[t])) { rv[t] = ov; ri[t] = oi; }
            }
            __syncthreads();
        }
        far = ri[0];
    }
}

// ---------------------------------------------------------------------------
// 2) Ball query: one wave32 per query point; ordered first-K via ballot.
// ---------------------------------------------------------------------------
__global__ __launch_bounds__(256) void qball_kernel(const float* __restrict__ xyz,
                                                    const float* __restrict__ new_xyz,
                                                    int* __restrict__ idxout) {
    __shared__ int buf[8 * Kc];
    const int wave = threadIdx.x >> 5;
    const int lane = threadIdx.x & 31;
    const int gs = blockIdx.x * 8 + wave;          // (b, s) flat
    const int b = gs / Sc;
    const float* X = xyz + (size_t)b * Nc * 3;

    const float cx = new_xyz[gs * 3 + 0];
    const float cy = new_xyz[gs * 3 + 1];
    const float cz = new_xyz[gs * 3 + 2];

    int cnt = 0;
    for (int chunk = 0; chunk < Nc / 32 && cnt < Kc; ++chunk) {
        const int n = chunk * 32 + lane;
        const float dx = X[n * 3 + 0] - cx;
        const float dy = X[n * 3 + 1] - cy;
        const float dz = X[n * 3 + 2] - cz;
        const bool pass = (dx * dx + dy * dy + dz * dz) <= R2c;
        const unsigned mask = (unsigned)__ballot(pass);
        if (pass) {
            const int slot = cnt + __popc(mask & ((1u << lane) - 1u));
            if (slot < Kc) buf[wave * Kc + slot] = n;
        }
        cnt += __popc(mask);
    }
    __syncthreads();
    const int first = buf[wave * Kc + 0];          // center always within radius
    idxout[gs * Kc + lane] = (lane < cnt) ? buf[wave * Kc + lane] : first;
}

// ---------------------------------------------------------------------------
// 3) Feature build: bf16, POINT-MAJOR featb[NP][96] (67..95 zero).
//    Each thread emits six contiguous 32B vector stores.
// ---------------------------------------------------------------------------
__global__ __launch_bounds__(256) void build_feat(const float* __restrict__ xyz,
                                                  const float* __restrict__ points,
                                                  const float* __restrict__ new_xyz,
                                                  const int* __restrict__ idx,
                                                  unsigned short* __restrict__ featb) {
    const int p = blockIdx.x * 256 + threadIdx.x;
    if (p >= NPc) return;
    const int b = p / (Sc * Kc);
    const int s = (p / Kc) % Sc;
    const int n = idx[p];
    const float* xp = xyz + ((size_t)(b * Nc + n)) * 3;
    const float* cp = new_xyz + ((size_t)(b * Sc + s)) * 3;
    const v16f* pv = (const v16f*)(points + ((size_t)(b * Nc + n)) * Dc); // 256B aligned
    const v16f P0 = pv[0], P1 = pv[1], P2 = pv[2], P3 = pv[3];
    const float g0 = xp[0] - cp[0], g1 = xp[1] - cp[1], g2 = xp[2] - cp[2];

    v16u chunk[6];
#pragma unroll
    for (int j = 0; j < 6; ++j)
#pragma unroll
        for (int e = 0; e < 16; ++e) {
            const int c = j * 16 + e;
            float v;
            if (c == 0)      v = g0;
            else if (c == 1) v = g1;
            else if (c == 2) v = g2;
            else if (c < 67) {
                const int q = c - 3;
                v = (q < 16) ? P0[q] : (q < 32) ? P1[q - 16]
                                     : (q < 48) ? P2[q - 32] : P3[q - 48];
            } else v = 0.0f;
            chunk[j][e] = f2bf(v);
        }
    unsigned short* dst = featb + (size_t)p * 96;
#pragma unroll
    for (int j = 0; j < 6; ++j) *(v16u*)(dst + j * 16) = chunk[j];
}

// ---------------------------------------------------------------------------
// 4) Weight prep: fp32 [M][Cin] -> bf16 bits [M][KP] (K zero-padded).
// ---------------------------------------------------------------------------
__global__ void prep_weights(const float* __restrict__ W, unsigned short* __restrict__ Wb,
                             int M, int Cin, int KP) {
    const int i = blockIdx.x * 256 + threadIdx.x;
    if (i >= M * KP) return;
    const int r = i / KP, c = i % KP;
    Wb[i] = f2bf((c < Cin) ? W[r * Cin + c] : 0.0f);
}

__global__ void zero_stats(float* __restrict__ stats) {
    const int i = blockIdx.x * 256 + threadIdx.x;
    if (i < 512) stats[i] = 0.0f;
}

// ---------------------------------------------------------------------------
// 5) GEMM + bias + per-channel sum/sumsq with bf16 WMMA 16x16x32.
//    IN_BF16: layer0 reads bf16 feat directly; else fp32 + fused BN+ReLU.
//    A-fragment = 2x ds_load_b128; B-fragment = contiguous global vectors.
//    Y output POINT-MAJOR [NP][M] -> v8f contiguous stores.
// ---------------------------------------------------------------------------
template <int M, int KP, bool IN_BF16>
__global__ __launch_bounds__(256) void gemm_bn_stats(const unsigned short* __restrict__ Wb,
                                                     const float* __restrict__ bias,
                                                     const void* __restrict__ Xin,
                                                     const float* __restrict__ aff_a,
                                                     const float* __restrict__ aff_c,
                                                     float* __restrict__ Yout,
                                                     float* __restrict__ gsum,
                                                     float* __restrict__ gsumsq) {
    __shared__ __align__(64) unsigned short sW[M * KP];
    __shared__ __align__(64) float sAff[IN_BF16 ? 1 : 2 * KP];
    __shared__ float sBias[M];
    __shared__ float ssum[M];
    __shared__ float ssq[M];

    const int tid = threadIdx.x;
    {   // cooperative 16B-vector load of the bf16 weight tile into LDS
        const uint4* src = (const uint4*)Wb;
        uint4* dst = (uint4*)sW;
        for (int i = tid; i < M * KP / 8; i += 256) dst[i] = src[i];
        for (int i = tid; i < M; i += 256) {
            sBias[i] = bias[i]; ssum[i] = 0.0f; ssq[i] = 0.0f;
        }
        if (!IN_BF16) {
            for (int i = tid; i < KP; i += 256) {
                sAff[i] = aff_a[i]; sAff[KP + i] = aff_c[i];
            }
        }
    }
    __syncthreads();

    const int wave  = tid >> 5;
    const int lane  = tid & 31;
    const int lhalf = lane >> 4;                    // 0: lanes 0-15, 1: 16-31
    const int l16   = lane & 15;
    const int p     = blockIdx.x * 128 + wave * 16 + l16;   // output column

    v8f acc[M / 16];
#pragma unroll
    for (int mi = 0; mi < M / 16; ++mi)
#pragma unroll
        for (int r = 0; r < 8; ++r) acc[mi][r] = 0.0f;

#pragma unroll
    for (int kb = 0; kb < KP / 32; ++kb) {
        // ---- B fragment: lane's column p, K = kb*32 + lhalf*16 + e, contiguous.
        v16bf bfrag;
        if (IN_BF16) {
            const unsigned short* xb = (const unsigned short*)Xin;
            bfrag = *(const v16bf*)(xb + (size_t)p * KP + kb * 32 + lhalf * 16);
        } else {
            const float* xf = (const float*)Xin;
            const v16f xv = *(const v16f*)(xf + (size_t)p * KP + kb * 32 + lhalf * 16);
            const v16f av = *(const v16f*)(sAff + kb * 32 + lhalf * 16);
            const v16f cv = *(const v16f*)(sAff + KP + kb * 32 + lhalf * 16);
            v16u bu;
#pragma unroll
            for (int e = 0; e < 16; ++e)
                bu[e] = f2bf(fmaxf(fmaf(av[e], xv[e], cv[e]), 0.0f));
            bfrag = __builtin_bit_cast(v16bf, bu);
        }

#pragma unroll
        for (int mi = 0; mi < M / 16; ++mi) {
            // ---- A fragment: one contiguous 16-element run per lane.
            //  lane<16 : K = {0..15} ; lane>=16 : K = {16..31}
            const uint4* ap = (const uint4*)(sW + (mi * 16 + l16) * KP + kb * 32 + lhalf * 16);
            struct U8 { uint4 a, b; };
            const U8 u8 = { ap[0], ap[1] };           // +0B and +16B
            const v16bf afrag = __builtin_bit_cast(v16bf, u8);
            acc[mi] = __builtin_amdgcn_wmma_f32_16x16x32_bf16(
                false, afrag, false, bfrag, (short)0, acc[mi], false, false);
        }
    }

    // ---- Epilogue: +bias, point-major contiguous v8f store, LDS stats.
#pragma unroll
    for (int mi = 0; mi < M / 16; ++mi) {
        const int row0 = mi * 16 + lhalf * 8;       // C/D layout: lane half -> +8
        v8f res;
#pragma unroll
        for (int r = 0; r < 8; ++r) res[r] = acc[mi][r] + sBias[row0 + r];
        *(v8f*)(Yout + (size_t)p * M + row0) = res;
#pragma unroll
        for (int r = 0; r < 8; ++r) {
            atomicAdd(&ssum[row0 + r], res[r]);
            atomicAdd(&ssq[row0 + r], res[r] * res[r]);
        }
    }
    __syncthreads();
    for (int i = tid; i < M; i += 256) {
        atomicAdd(&gsum[i], ssum[i]);
        atomicAdd(&gsumsq[i], ssq[i]);
    }
}

// ---------------------------------------------------------------------------
// 6) BN finalize: a = g * rsqrt(var + eps), c = be - mu * a.
// ---------------------------------------------------------------------------
__global__ void finalize_bn(const float* __restrict__ gsum, const float* __restrict__ gsumsq,
                            const float* __restrict__ gamma, const float* __restrict__ beta,
                            float* __restrict__ a, float* __restrict__ c, int M) {
    const int i = blockIdx.x * blockDim.x + threadIdx.x;
    if (i >= M) return;
    const float invN = 1.0f / (float)NPc;
    const float mu  = gsum[i] * invN;
    const float var = gsumsq[i] * invN - mu * mu;
    const float sc  = gamma[i] * rsqrtf(var + 1e-5f);
    a[i] = sc;
    c[i] = beta[i] - mu * sc;
}

// ---------------------------------------------------------------------------
// 7) Fused BN + ReLU + max over K; Y2 point-major [NP][128].
//    Thread map (b, s, ch) ch-fastest keeps the 128MB of reads coalesced.
// ---------------------------------------------------------------------------
__global__ __launch_bounds__(256) void max_kernel(const float* __restrict__ Y2,
                                                  const float* __restrict__ a2,
                                                  const float* __restrict__ c2,
                                                  float* __restrict__ out) {
    const int i = blockIdx.x * 256 + threadIdx.x;   // i = ((b*S + s)*128 + ch)
    if (i >= Bc * Sc * 128) return;
    const int ch = i & 127;
    const int s = (i >> 7) % Sc;
    const int b = i / (Sc * 128);
    const size_t base = ((size_t)(b * Sc + s) * Kc) * 128 + ch;
    const float av = a2[ch], cv = c2[ch];
    float m = 0.0f;                                  // relu outputs >= 0
#pragma unroll
    for (int k = 0; k < Kc; ++k) {
        const float y = Y2[base + (size_t)k * 128];
        m = fmaxf(m, fmaxf(fmaf(av, y, cv), 0.0f));
    }
    out[Bc * Sc * 3 + ((size_t)(b * 128 + ch) * Sc + s)] = m;
}

// ---------------------------------------------------------------------------
// Launcher
// ---------------------------------------------------------------------------
extern "C" void kernel_launch(void* const* d_in, const int* in_sizes, int n_in,
                              void* d_out, int out_size, void* d_ws, size_t ws_size,
                              hipStream_t stream) {
    const float* xyz    = (const float*)d_in[0];
    const float* points = (const float*)d_in[1];
    const float* w0 = (const float*)d_in[2],  *b0 = (const float*)d_in[3];
    const float* g0 = (const float*)d_in[4],  *be0 = (const float*)d_in[5];
    const float* w1 = (const float*)d_in[6],  *b1 = (const float*)d_in[7];
    const float* g1 = (const float*)d_in[8],  *be1 = (const float*)d_in[9];
    const float* w2 = (const float*)d_in[10], *b2 = (const float*)d_in[11];
    const float* g2 = (const float*)d_in[12], *be2 = (const float*)d_in[13];

    float* out = (float*)d_out;
    char* ws = (char*)d_ws;

    size_t off = 0;
    auto take = [&](size_t bytes) { size_t o = off; off = (off + bytes + 255) & ~(size_t)255; return o; };
    int*            fps_idx = (int*)(ws + take((size_t)Bc * Sc * 4));
    int*            qidx    = (int*)(ws + take((size_t)Bc * Sc * Kc * 4));
    unsigned short* Wb0     = (unsigned short*)(ws + take((size_t)64 * 96 * 2));
    unsigned short* Wb1     = (unsigned short*)(ws + take((size_t)64 * 64 * 2));
    unsigned short* Wb2     = (unsigned short*)(ws + take((size_t)128 * 64 * 2));
    float*          stats   = (float*)(ws + take(512 * 4));
    float*          affine  = (float*)(ws + take(512 * 4));
    unsigned short* featb   = (unsigned short*)(ws + take((size_t)NPc * 96 * 2)); // bf16 [NP][96]
    float*          Y0      = (float*)(ws + take((size_t)NPc * 64 * 4));          // fp32 [NP][64]
    float*          Y1      = (float*)(ws + take((size_t)NPc * 64 * 4));
    float*          Y2      = (float*)(ws + take((size_t)NPc * 128 * 4));         // fp32 [NP][128]

    float* gsum0 = stats,       *gsq0 = stats + 64;
    float* gsum1 = stats + 128, *gsq1 = stats + 192;
    float* gsum2 = stats + 256, *gsq2 = stats + 384;
    float* a0 = affine,       *c0 = affine + 64;
    float* a1 = affine + 128, *c1 = affine + 192;
    float* a2 = affine + 256, *c2 = affine + 384;

    float* new_xyz = out;                     // first B*S*3 floats of output

    fps_kernel<<<Bc, 256, 0, stream>>>(xyz, fps_idx, new_xyz);
    qball_kernel<<<(Bc * Sc) / 8, 256, 0, stream>>>(xyz, new_xyz, qidx);
    build_feat<<<(NPc + 255) / 256, 256, 0, stream>>>(xyz, points, new_xyz, qidx, featb);

    zero_stats<<<2, 256, 0, stream>>>(stats);
    prep_weights<<<(64 * 96 + 255) / 256, 256, 0, stream>>>(w0, Wb0, 64, 67, 96);
    prep_weights<<<(64 * 64 + 255) / 256, 256, 0, stream>>>(w1, Wb1, 64, 64, 64);
    prep_weights<<<(128 * 64 + 255) / 256, 256, 0, stream>>>(w2, Wb2, 128, 64, 64);

    const int gemm_blocks = NPc / 128;        // 2048
    gemm_bn_stats<64, 96, true><<<gemm_blocks, 256, 0, stream>>>(
        Wb0, b0, featb, nullptr, nullptr, Y0, gsum0, gsq0);
    finalize_bn<<<1, 64, 0, stream>>>(gsum0, gsq0, g0, be0, a0, c0, 64);

    gemm_bn_stats<64, 64, false><<<gemm_blocks, 256, 0, stream>>>(
        Wb1, b1, Y0, a0, c0, Y1, gsum1, gsq1);
    finalize_bn<<<1, 64, 0, stream>>>(gsum1, gsq1, g1, be1, a1, c1, 64);

    gemm_bn_stats<128, 64, false><<<gemm_blocks, 256, 0, stream>>>(
        Wb2, b2, Y1, a1, c1, Y2, gsum2, gsq2);
    finalize_bn<<<1, 128, 0, stream>>>(gsum2, gsq2, g2, be2, a2, c2, 128);

    max_kernel<<<(Bc * Sc * 128) / 256, 256, 0, stream>>>(Y2, a2, c2, out);
}
